// STTransformerWithDeformableAttention_57818849738954
// MI455X (gfx1250) — matrix operand order repaired and checked by reading, not compiled, not run on hardware
//
#include <hip/hip_runtime.h>
#include <hip/hip_bf16.h>

typedef __attribute__((ext_vector_type(16))) _Float16 v16h;
typedef __attribute__((ext_vector_type(8)))  float    v8f;

#define NB    8        // batch
#define LQ    4096     // queries
#define CDIM  256      // channels (== GEMM K)
#define NHEAD 8
#define DH    32       // channels per head (== wave32 lanes)
#define NLVL  4
#define NPT   4
#define MROWS (NB * LQ)   // 32768 GEMM rows
#define KCHUNKS (CDIM / 32)

// ---------------------------------------------------------------------------
// Pack W[256, Ncols] (f32, row-major) into WMMA B fragments with f16 hi/lo
// split, in the exact per-lane layout v_wmma_f32_16x16x32_f16 consumes
// (ISA 7.12.2: N on lanes, K packed 2/VGPR; lanes 0-15 K {0..7,16..23}, 16-31 K {8..15,24..31}).
//   pack[(((tile*8 + chunk)*2 + s)*32 + lane)] = v16h fragment (s=0: hi, s=1: lo)
// One thread per (tile, chunk, lane). Runs once per launch; cost is noise.
// ---------------------------------------------------------------------------
__global__ __launch_bounds__(256)
void pack_w_kernel(const float* __restrict__ W, v16h* __restrict__ outp, int Ncols)
{
    const int tid   = blockIdx.x * 256 + threadIdx.x;
    const int lane  = tid & 31;
    const int chunk = (tid >> 5) & (KCHUNKS - 1);
    const int tile  = tid >> 8;
    if (tile * 16 >= Ncols) return;

    const int half = (lane < 16) ? 0 : 1;
    const int lidx = lane & 15;
    const int ncol = tile * 16 + lidx;
    const int kb   = chunk * 32 + (half ? 8 : 0);

    v16h b1, b2;
#pragma unroll
    for (int e = 0; e < 16; ++e) {
        const float f = W[(kb + e + (e & 8)) * Ncols + ncol];
        const _Float16 h = (_Float16)f;
        b1[e] = h;
        b2[e] = (_Float16)(f - (float)h);
    }
    const size_t base = ((size_t)(tile * KCHUNKS + chunk) * 2) * 32 + lane;
    outp[base]      = b1;
    outp[base + 32] = b2;
}

// ---------------------------------------------------------------------------
// WMMA GEMM:  Y[M, Ncols] = rowmap(A)[M, 256] @ W[256, Ncols] + bias
//   a_mode==1 : A row r=(n*LQ+lq) is read from query[lq, n, :]  (transpose in)
//   c_mode==1 : Y row r=(n*LQ+lq) is written to out[lq, n, :]   (transpose out)
// W comes pre-packed as split-f16 B fragments -> two coalesced b128 loads per
// fragment, no conversion VALU in the hot loop. A is loaded as 4x b128 per
// chunk and hi/lo split on the fly (A1B1 + A1B2 + A2B1 ~ fp32 accuracy).
// One wave owns a 16x64 tile; all control flow wave-uniform (EXEC all-ones).
// ---------------------------------------------------------------------------
__global__ __launch_bounds__(256)
void wmma_gemm_kernel(const float* __restrict__ A, const v16h* __restrict__ Wpack,
                      const float* __restrict__ bias, float* __restrict__ Y,
                      int Ncols, int a_mode, int c_mode)
{
    const int lane  = threadIdx.x & 31;
    const int wave  = threadIdx.x >> 5;
    const int mtile = blockIdx.x * 8 + wave;     // 16-row tile index
    const int half  = (lane < 16) ? 0 : 1;       // half-wave select
    const int lidx  = lane & 15;

    // Per-lane A row pointer (M = mtile*16 + lidx), contiguous over K.
    const int mrow = mtile * 16 + lidx;
    const float* arow;
    if (a_mode == 1) {
        const int n = mrow / LQ, lq = mrow % LQ;             // r = n*LQ + lq
        arow = A + ((size_t)lq * NB + n) * CDIM;             // query[lq, n, :]
    } else {
        arow = A + (size_t)mrow * CDIM;
    }

    v8f acc[4];
#pragma unroll
    for (int t = 0; t < 4; ++t) acc[t] = 0.0f;

    // A 16x32 f16 layout: lanes 0-15 hold K {0..7, 16..23},
    // lanes 16-31 hold K {8..15, 24..31}.
    const int kbaseA = half ? 8 : 0;

    for (int kchunk = 0; kchunk < KCHUNKS; ++kchunk) {
        const int kc = kchunk * 32;

        // ---- A fragment: 4x float4 (32B-aligned) + hi/lo f16 split ----
        float fa[16];
        *(float4*)(fa + 0)  = *(const float4*)(arow + kc + kbaseA + 0);
        *(float4*)(fa + 4)  = *(const float4*)(arow + kc + kbaseA + 4);
        *(float4*)(fa + 8)  = *(const float4*)(arow + kc + kbaseA + 16);
        *(float4*)(fa + 12) = *(const float4*)(arow + kc + kbaseA + 20);

        v16h a1, a2;
#pragma unroll
        for (int e = 0; e < 16; ++e) {
            const _Float16 h = (_Float16)fa[e];
            a1[e] = h;
            a2[e] = (_Float16)(fa[e] - (float)h);
        }

        // ---- 4 column tiles: pre-packed B fragments, 3-term split WMMA ----
#pragma unroll
        for (int t = 0; t < 4; ++t) {
            const size_t fb =
                ((size_t)((blockIdx.y * 4 + t) * KCHUNKS + kchunk) * 2) * 32 + lane;
            const v16h b1 = Wpack[fb];        // two global_load_b128, coalesced
            const v16h b2 = Wpack[fb + 32];
            acc[t] = __builtin_amdgcn_wmma_f32_16x16x32_f16(
                false, a1, false, b1, (short)0, acc[t], false, false);
            acc[t] = __builtin_amdgcn_wmma_f32_16x16x32_f16(
                false, a1, false, b2, (short)0, acc[t], false, false);
            acc[t] = __builtin_amdgcn_wmma_f32_16x16x32_f16(
                false, a2, false, b1, (short)0, acc[t], false, false);
        }
    }

    // C/D layout: VGPR v -> M = v (lanes 0-15) / v+8 (lanes 16-31), N = lane%16
    const int mbase = mtile * 16 + half * 8;
#pragma unroll
    for (int t = 0; t < 4; ++t) {
        const int ncol = (blockIdx.y * 4 + t) * 16 + lidx;
        const float bv = bias[ncol];
#pragma unroll
        for (int v = 0; v < 8; ++v) {
            const int m = mbase + v;
            const float val = acc[t][v] + bv;
            if (c_mode == 1) {
                const int n = m / LQ, lq = m % LQ;
                Y[((size_t)lq * NB + n) * Ncols + ncol] = val;   // out[lq, n, c]
            } else {
                Y[(size_t)m * Ncols + ncol] = val;
            }
        }
    }
}

// ---------------------------------------------------------------------------
// Fused softmax + bilinear sampling + attention-weighted aggregation.
// Block = 256 threads = 8 waves; blockIdx.x = r = n*LQ + lq; wave = head.
// Lane = channel within head -> every corner fetch is one coalesced 128B line
// from the (L2-resident, 108 MB) feature map. All coordinate math and
// branches are wave-uniform.
// ---------------------------------------------------------------------------
__global__ __launch_bounds__(256)
void deform_sample_kernel(const float* __restrict__ off,      // (MROWS, 256)
                          const float* __restrict__ logits,   // (MROWS, 128)
                          const float* __restrict__ refp,     // (N, LQ, 4, 2)
                          const float* __restrict__ fmaps,    // (Ltot, N, C)
                          float* __restrict__ agg)            // (MROWS, 256)
{
    const int lane = threadIdx.x & 31;
    const int h    = threadIdx.x >> 5;
    const int r    = blockIdx.x;
    const int n    = r >> 12;                    // r / LQ (LQ = 4096)

    // softmax over the 16 (level, point) logits of this head (wave-uniform)
    const float* lg = logits + (size_t)r * (NHEAD * NLVL * NPT) + h * (NLVL * NPT);
    float w[16];
    float mx = -3.0e38f;
#pragma unroll
    for (int i = 0; i < 16; ++i) mx = fmaxf(mx, lg[i]);
    float den = 0.0f;
#pragma unroll
    for (int i = 0; i < 16; ++i) { w[i] = __expf(lg[i] - mx); den += w[i]; }
    const float inv = 1.0f / den;

    const float* offr = off + (size_t)r * 256 + h * (NLVL * NPT * 2);
    const float* rp   = refp + (size_t)r * (NLVL * 2);

    const int WW[4] = {100, 50, 25, 13};
    const int HH[4] = {100, 50, 25, 13};
    const int ST[4] = {0, 10000, 12500, 13125};   // level row starts

    const int cofs = n * CDIM + h * DH + lane;    // channel offset into fmaps row
    float acc = 0.0f;

#pragma unroll
    for (int l = 0; l < NLVL; ++l) {
        const int   Wl = WW[l], Hl = HH[l], st = ST[l];
        const float fW = (float)Wl, fH = (float)Hl;
        const float rx = rp[l * 2 + 0], ry = rp[l * 2 + 1];
#pragma unroll
        for (int p = 0; p < NPT; ++p) {
            const float ox = offr[(l * NPT + p) * 2 + 0];
            const float oy = offr[(l * NPT + p) * 2 + 1];
            // x = loc_x*W - 0.5 with loc_x = rx + ox/W  (grid_sample algebra)
            const float x = rx * fW + ox - 0.5f;
            const float y = ry * fH + oy - 0.5f;
            const float x0f = floorf(x), y0f = floorf(y);
            const float wx = x - x0f,   wy = y - y0f;
            const int   x0 = (int)x0f,  y0 = (int)y0f;
            const float aw = w[l * NPT + p] * inv;

            const bool xin0 = (x0 >= 0)     && (x0 < Wl);
            const bool xin1 = (x0 + 1 >= 0) && (x0 + 1 < Wl);
            const bool yin0 = (y0 >= 0)     && (y0 < Hl);
            const bool yin1 = (y0 + 1 >= 0) && (y0 + 1 < Hl);

            float v00 = 0.f, v01 = 0.f, v10 = 0.f, v11 = 0.f;
            if (yin0) {
                const size_t rowb = (size_t)(st + y0 * Wl) * (NB * CDIM) + cofs;
                if (xin0) v00 = fmaps[rowb + (size_t)x0 * (NB * CDIM)];
                if (xin1) v01 = fmaps[rowb + (size_t)(x0 + 1) * (NB * CDIM)];
            }
            if (yin1) {
                const size_t rowb = (size_t)(st + (y0 + 1) * Wl) * (NB * CDIM) + cofs;
                if (xin0) v10 = fmaps[rowb + (size_t)x0 * (NB * CDIM)];
                if (xin1) v11 = fmaps[rowb + (size_t)(x0 + 1) * (NB * CDIM)];
            }
            const float top = v00 + wx * (v01 - v00);
            const float bot = v10 + wx * (v11 - v10);
            acc += aw * (top + wy * (bot - top));
        }
    }
    agg[(size_t)r * CDIM + h * DH + lane] = acc;
}

// ---------------------------------------------------------------------------
extern "C" void kernel_launch(void* const* d_in, const int* in_sizes, int n_in,
                              void* d_out, int out_size, void* d_ws, size_t ws_size,
                              hipStream_t stream)
{
    const float* query  = (const float*)d_in[0];  // (LQ, N, C)
    const float* refp   = (const float*)d_in[1];  // (N, LQ, 4, 2)
    const float* fmaps  = (const float*)d_in[2];  // (13294, N, C)
    const float* W_off  = (const float*)d_in[4];  // (256, 256)
    const float* b_off  = (const float*)d_in[5];
    const float* W_attn = (const float*)d_in[6];  // (256, 128)
    const float* b_attn = (const float*)d_in[7];
    const float* W_out  = (const float*)d_in[8];  // (256, 256)
    const float* b_out  = (const float*)d_in[9];
    float* out = (float*)d_out;                   // (LQ, N, C)

    // workspace: off (33.5 MB) | logits (16.8 MB) | agg (33.5 MB) | packed W
    float* off    = (float*)d_ws;
    float* logits = off + (size_t)MROWS * 256;
    float* agg    = logits + (size_t)MROWS * 128;
    v16h*  p_off  = (v16h*)(agg + (size_t)MROWS * 256);          // 16 tiles: 256 KB
    v16h*  p_attn = p_off  + (size_t)16 * KCHUNKS * 2 * 32;      //  8 tiles: 128 KB
    v16h*  p_out  = p_attn + (size_t)8  * KCHUNKS * 2 * 32;      // 16 tiles: 256 KB

    const dim3 blk(256);
    // pack the three weight matrices into WMMA B fragments (once per launch)
    pack_w_kernel<<<dim3((256 / 16) * 256 / 256), blk, 0, stream>>>(W_off,  p_off,  256);
    pack_w_kernel<<<dim3((128 / 16) * 256 / 256), blk, 0, stream>>>(W_attn, p_attn, 128);
    pack_w_kernel<<<dim3((256 / 16) * 256 / 256), blk, 0, stream>>>(W_out,  p_out,  256);

    // sampling offsets:  (N*Lq, 256) = q @ W_off + b_off   (A transposed in)
    wmma_gemm_kernel<<<dim3(MROWS / 128, 256 / 64), blk, 0, stream>>>(
        query, p_off, b_off, off, 256, /*a_mode=*/1, /*c_mode=*/0);
    // attention logits:  (N*Lq, 128) = q @ W_attn + b_attn
    wmma_gemm_kernel<<<dim3(MROWS / 128, 128 / 64), blk, 0, stream>>>(
        query, p_attn, b_attn, logits, 128, /*a_mode=*/1, /*c_mode=*/0);
    // fused softmax + deformable bilinear sampling + weighted aggregation
    deform_sample_kernel<<<dim3(MROWS), blk, 0, stream>>>(
        off, logits, refp, fmaps, agg);
    // output projection with transpose-back store: out[lq, n, :] = agg @ W_out
    wmma_gemm_kernel<<<dim3(MROWS / 128, 256 / 64), blk, 0, stream>>>(
        agg, p_out, b_out, out, 256, /*a_mode=*/0, /*c_mode=*/1);
}
